// PPOTrustGNN_197568495999
// MI455X (gfx1250) — matrix operand order repaired and checked by reading, not compiled, not run on hardware
//
#include <hip/hip_runtime.h>
#include <hip/hip_bf16.h>
#include <math.h>

// ---------------------------------------------------------------------------
// PPO-Trust hetero-GAT forward for MI455X (gfx1250, wave32, WMMA).
// Dense matmuls: v_wmma_f32_16x16x32_f16 tiled GEMM with async global->LDS
// tile staging (ASYNCcnt) and LDS-staged coalesced epilogue.
// ---------------------------------------------------------------------------

typedef __attribute__((ext_vector_type(16))) _Float16 v16h;
typedef __attribute__((ext_vector_type(8)))  float    v8f;

#define NA_ 10000
#define NT_ 20000
#define E_  100000

// ---------------- monotone float<->uint for atomic max --------------------
__device__ __forceinline__ unsigned f2mono(float f) {
  unsigned b = __float_as_uint(f);
  return (b & 0x80000000u) ? ~b : (b | 0x80000000u);
}
__device__ __forceinline__ float mono2f(unsigned u) {
  unsigned b = (u & 0x80000000u) ? (u ^ 0x80000000u) : ~u;
  return __uint_as_float(b);
}

// issue one 16B async global->LDS copy (per-lane, EXEC-masked)
__device__ __forceinline__ void async_copy16(const float* gp, const float* lp) {
  unsigned lo = (unsigned)(size_t)lp;     // low 32 bits of LDS flat addr = offset
  asm volatile("global_load_async_to_lds_b128 %0, %1, off"
               :: "v"(lo), "v"(gp) : "memory");
}

// ---------------------------------------------------------------------------
// WMMA GEMM: C[M,N] = act(A[M,K] @ B[K,N] + bias), fp32 in/out, f16 compute.
// K <= 64 (zero padded; K, N multiples of 4). Block = 256 threads = 8 waves;
// tile 64(M) x 64(N); each wave computes 16x32 (A fragment reused over two
// 16x16x32 WMMA column tiles, two K steps -> 4 WMMA per wave).
// ---------------------------------------------------------------------------
__global__ __launch_bounds__(256) void gemm_wmma_kernel(
    const float* __restrict__ A, const float* __restrict__ B,
    const float* __restrict__ bias, float* __restrict__ C,
    int M, int N, int K, int act)
{
  __shared__ __align__(16) float smem[2 * 64 * 68];
  float* Af = smem;             // [64][68] A tile, row-major f32
  float* Bf = smem + 64 * 68;   // [64][68] B tile, row-major f32: Bf[k][n]

  const int tid  = threadIdx.x;
  const int wave = tid >> 5;
  const int lane = tid & 31;
  const int bm   = blockIdx.x * 64;
  const int bn   = blockIdx.y * 64;

  // Zero both padded tiles so OOB rows/cols and K padding contribute 0.
  for (int i = tid; i < 2176; i += 256)
    ((float4*)smem)[i] = make_float4(0.f, 0.f, 0.f, 0.f);
  __syncthreads();

  // Async-stage A tile rows (K multiple of 4 -> whole 16B chunks).
  const int kq = K >> 2;
  for (int i = tid; i < 64 * kq; i += 256) {
    int r = i / kq, cq = i - r * kq;
    int gr = bm + r;
    if (gr < M)
      async_copy16(A + (size_t)gr * K + cq * 4, Af + r * 68 + cq * 4);
  }
  // Async-stage B tile rows (N multiple of 4 -> whole 16B chunks).
  for (int i = tid; i < 64 * 16; i += 256) {
    int k = i >> 4, cq = i & 15;
    int gc = bn + cq * 4;
    if (k < K && gc < N)
      async_copy16(B + (size_t)k * N + gc, Bf + k * 68 + cq * 4);
  }
  asm volatile("s_wait_asynccnt 0x0" ::: "memory");
  __syncthreads();

  const int m0   = (wave & 3) << 4;   // row sub-tile (0..48)
  const int nc   = (wave >> 2) << 5;  // col group base (0 or 32)
  const int lo16 = lane & 15;
  const int hi   = lane >> 4;

  v8f acc0 = {}, acc1 = {};
#pragma unroll
  for (int ks = 0; ks < 2; ++ks) {
    const int kb = ks * 32;
    v16h a, b0, b1;
    const float* Ar = Af + (m0 + lo16) * 68;
#pragma unroll
    for (int j = 0; j < 8; ++j) {
      // A fragment: lanes 0-15 K=0..15, lanes 16-31 K=16..31
      int ka = kb + 2 * j + 16 * hi;
      a[2 * j]     = (_Float16)Ar[ka];
      a[2 * j + 1] = (_Float16)Ar[ka + 1];
      // B fragment: lanes 0-15 K=0..15, lanes 16-31 K=16..31
      int kbb = kb + 2 * j + 16 * hi;
      b0[2 * j]     = (_Float16)Bf[kbb * 68 + nc + lo16];
      b0[2 * j + 1] = (_Float16)Bf[(kbb + 1) * 68 + nc + lo16];
      b1[2 * j]     = (_Float16)Bf[kbb * 68 + nc + 16 + lo16];
      b1[2 * j + 1] = (_Float16)Bf[(kbb + 1) * 68 + nc + 16 + lo16];
    }
    acc0 = __builtin_amdgcn_wmma_f32_16x16x32_f16(
        false, a, false, b0, (short)0, acc0, false, false);
    acc1 = __builtin_amdgcn_wmma_f32_16x16x32_f16(
        false, a, false, b1, (short)0, acc1, false, false);
  }

  // Stage C tile through LDS (reuse A region), then coalesced b128 stores.
  __syncthreads();
  float* Cs = smem;   // [64][68]
#pragma unroll
  for (int r = 0; r < 8; ++r) {
    Cs[(m0 + r + 8 * hi) * 68 + nc + lo16]      = acc0[r];
    Cs[(m0 + r + 8 * hi) * 68 + nc + 16 + lo16] = acc1[r];
  }
  __syncthreads();
  for (int i = tid; i < 64 * 16; i += 256) {
    int r = i >> 4, cq = i & 15;
    int gr = bm + r, gc = bn + cq * 4;
    if (gr < M && gc < N) {
      float4 v = *(float4*)&Cs[r * 68 + cq * 4];
      if (bias) {
        v.x += bias[gc];     v.y += bias[gc + 1];
        v.z += bias[gc + 2]; v.w += bias[gc + 3];
      }
      if (act) {
        v.x = fmaxf(v.x, 0.f); v.y = fmaxf(v.y, 0.f);
        v.z = fmaxf(v.z, 0.f); v.w = fmaxf(v.w, 0.f);
      }
      *(float4*)&C[(size_t)gr * N + gc] = v;
    }
  }
}

// ---------------------------------------------------------------------------
// Per-(node,head) attention projection: out[n,h] = dot(Hm[n,h,:64], att[h,:64])
// ---------------------------------------------------------------------------
__global__ void head_dot_kernel(const float* __restrict__ Hm,
                                const float* __restrict__ att,
                                float* __restrict__ out, int Nn, int H)
{
  int i = blockIdx.x * blockDim.x + threadIdx.x;
  if (i >= Nn * H) return;
  int n = i / H, h = i - n * H;
  const float* hp = Hm + (size_t)n * H * 64 + h * 64;
  const float* ap = att + h * 64;
  float s = 0.f;
#pragma unroll 4
  for (int d = 0; d < 64; ++d) s += hp[d] * ap[d];
  out[i] = s;
}

// per-edge score (leaky relu 0.2) + segment max via monotone-uint atomicMax
__global__ void edge_score_max_kernel(const int* __restrict__ src,
                                      const int* __restrict__ dst,
                                      const float* __restrict__ asrc,
                                      const float* __restrict__ adst,
                                      float* __restrict__ sbuf,
                                      unsigned* __restrict__ amax, int H)
{
  int i = blockIdx.x * blockDim.x + threadIdx.x;
  if (i >= E_ * H) return;
  int e = i / H, h = i - e * H;
  float s = asrc[src[e] * H + h] + adst[dst[e] * H + h];
  s = (s > 0.f) ? s : 0.2f * s;
  sbuf[i] = s;
  atomicMax(&amax[dst[e] * H + h], f2mono(s));
}

// per-edge exp(a - amax[dst]) + segment sum of denominators
__global__ void edge_exp_kernel(const int* __restrict__ dst,
                                float* __restrict__ sbuf,
                                const unsigned* __restrict__ amax,
                                float* __restrict__ denom, int H)
{
  int i = blockIdx.x * blockDim.x + threadIdx.x;
  if (i >= E_ * H) return;
  int e = i / H, h = i - e * H;
  int di = dst[e] * H + h;
  float ex = __expf(sbuf[i] - mono2f(amax[di]));
  sbuf[i] = ex;
  atomicAdd(&denom[di], ex);
}

// scatter alpha * h_src into dst accumulator (float4 granularity)
__global__ void edge_scatter_kernel(const int* __restrict__ src,
                                    const int* __restrict__ dst,
                                    const float* __restrict__ sbuf,
                                    const float* __restrict__ denom,
                                    const float* __restrict__ HS,
                                    float* __restrict__ acc, int H)
{
  int i = blockIdx.x * blockDim.x + threadIdx.x;
  int total = E_ * H * 16;
  if (i >= total) return;
  int e   = i / (H * 16);
  int rem = i - e * (H * 16);
  int h   = rem >> 4;
  int q   = rem & 15;
  int s   = src[e], d = dst[e];
  float alpha = sbuf[e * H + h] / (denom[d * H + h] + 1e-16f);
  const float4 v = *(const float4*)(HS + (size_t)s * H * 64 + h * 64 + 4 * q);
  float* ap = acc + (size_t)d * H * 64 + h * 64 + 4 * q;
  atomicAdd(ap + 0, v.x * alpha);
  atomicAdd(ap + 1, v.y * alpha);
  atomicAdd(ap + 2, v.z * alpha);
  atomicAdd(ap + 3, v.w * alpha);
}

// y = relu(mean_heads(acc) + b1 + b2); LDS-reduced batch stats -> global atomics
__global__ __launch_bounds__(256) void gat_finalize_kernel(
    const float* __restrict__ acc, const float* __restrict__ b1,
    const float* __restrict__ b2, float* __restrict__ Y,
    float* __restrict__ ssum, float* __restrict__ ssq, int Nn, int H)
{
  __shared__ float ls[64], lq[64];
  int tid = threadIdx.x;
  if (tid < 64) { ls[tid] = 0.f; lq[tid] = 0.f; }
  __syncthreads();
  int d = tid & 63;
  int n = blockIdx.x * 4 + (tid >> 6);
  if (n < Nn) {
    float s = 0.f;
    for (int h = 0; h < H; ++h) s += acc[(size_t)n * H * 64 + h * 64 + d];
    float y = s / (float)H + b1[d] + b2[d];
    y = fmaxf(y, 0.f);
    Y[(size_t)n * 64 + d] = y;
    atomicAdd(&ls[d], y);
    atomicAdd(&lq[d], y * y);
  }
  __syncthreads();
  if (tid < 64) { atomicAdd(&ssum[tid], ls[tid]); atomicAdd(&ssq[tid], lq[tid]); }
}

// training-mode batchnorm apply (+ optional residual)
__global__ void bn_apply_kernel(const float* __restrict__ Y,
                                const float* __restrict__ ssum,
                                const float* __restrict__ ssq,
                                const float* __restrict__ gamma,
                                const float* __restrict__ beta,
                                const float* __restrict__ resid,
                                float* __restrict__ Xout, int Nn)
{
  int i = blockIdx.x * blockDim.x + threadIdx.x;
  if (i >= Nn * 64) return;
  int d = i & 63;
  float inv_n = 1.f / (float)Nn;
  float m = ssum[d] * inv_n;
  float var = ssq[d] * inv_n - m * m;
  float z = (Y[i] - m) * rsqrtf(var + 1e-5f) * gamma[d] + beta[d];
  if (resid) z += resid[i];
  Xout[i] = z;
}

// wave-per-row layernorm + relu (R = 32 or 64), in-place safe
__global__ __launch_bounds__(256) void ln_relu_kernel(
    const float* __restrict__ X, const float* __restrict__ g,
    const float* __restrict__ b, float* __restrict__ Y, int Nn, int R)
{
  int wave = threadIdx.x >> 5, lane = threadIdx.x & 31;
  int n = blockIdx.x * 8 + wave;
  if (n >= Nn) return;
  const float* x = X + (size_t)n * R;
  float v0 = x[lane];
  float v1 = (R == 64) ? x[lane + 32] : 0.f;
  float s = v0 + v1;
#pragma unroll
  for (int o = 16; o; o >>= 1) s += __shfl_xor(s, o);
  float mean = s / (float)R;
  float d0 = v0 - mean, d1 = (R == 64) ? (v1 - mean) : 0.f;
  float q = d0 * d0 + d1 * d1;
#pragma unroll
  for (int o = 16; o; o >>= 1) q += __shfl_xor(q, o);
  float inv = rsqrtf(q / (float)R + 1e-5f);
  Y[(size_t)n * R + lane] = fmaxf(d0 * inv * g[lane] + b[lane], 0.f);
  if (R == 64)
    Y[(size_t)n * R + lane + 32] = fmaxf(d1 * inv * g[lane + 32] + b[lane + 32], 0.f);
}

// policy tail: out[n,col] = softplus(h[n,:32] @ W2 + b2) + 1
__global__ __launch_bounds__(256) void policy_out_kernel(
    const float* __restrict__ Hm, const float* __restrict__ W2,
    const float* __restrict__ b2, float* __restrict__ out, int Nn, int col)
{
  int wave = threadIdx.x >> 5, lane = threadIdx.x & 31;
  int n = blockIdx.x * 8 + wave;
  if (n >= Nn) return;
  float v = Hm[(size_t)n * 32 + lane] * W2[lane];
#pragma unroll
  for (int o = 16; o; o >>= 1) v += __shfl_xor(v, o);
  if (lane == 0) {
    float xv = v + b2[0];
    float sp = (xv > 20.f) ? xv : log1pf(__expf(xv));
    out[(size_t)n * 5 + col] = sp + 1.f;
  }
}

// value tail: v = h[n,:16] @ W4 + b4; v = v*(1-reg) + reg*tanh(v)
__global__ __launch_bounds__(256) void value_out_kernel(
    const float* __restrict__ Hm, const float* __restrict__ W4,
    const float* __restrict__ b4, const float* __restrict__ regp,
    float* __restrict__ out, int Nn)
{
  int wave = threadIdx.x >> 5, lane = threadIdx.x & 31;
  int n = blockIdx.x * 8 + wave;
  if (n >= Nn) return;
  float v = (lane < 16) ? Hm[(size_t)n * 16 + lane] * W4[lane] : 0.f;
#pragma unroll
  for (int o = 16; o; o >>= 1) v += __shfl_xor(v, o);
  if (lane == 0) {
    float vv = v + b4[0];
    float reg = regp[0];
    out[(size_t)n * 5 + 4] = vv * (1.f - reg) + reg * tanhf(vv);
  }
}

// ---------------------------------------------------------------------------
// Host orchestration
// ---------------------------------------------------------------------------
extern "C" void kernel_launch(void* const* d_in, const int* in_sizes, int n_in,
                              void* d_out, int out_size, void* d_ws, size_t ws_size,
                              hipStream_t stream)
{
  (void)in_sizes; (void)n_in; (void)out_size; (void)ws_size;

  auto P = [&](int i) { return (const float*)d_in[i]; };
  auto I = [&](int i) { return (const int*)d_in[i]; };

  // ---- param leaf indices (jax pytree: dict keys sorted, lists in order) ---
  // mlp2:   W1=0 W2=1 b1=2 b2=3 ln_b=4 ln_g=5
  // valmlp: W1=0 W2=1 W3=2 W4=3 b1=4 b2=5 b3=6 b4=7 ln1_b=8 ln1_g=9 ln2_b=10 ln2_g=11
  const int AG_CA = 10, AG_CB = 16, AG_V = 22, AG_VA = 34, AG_VB = 40;
  const int BN0 = 46;    // +l*4: agent beta, agent gamma, track beta, track gamma
  const int CONV0 = 58;  // +l*16 + slot*4: W, att_dst, att_src, bias
                         // slot order (sorted): at1=0, at2=1, ta1=2, ta2=3
  const int EMB_AW = 106, EMB_AB = 107, EMB_TW = 108, EMB_TB = 109;
  const int TR_CA = 110, TR_CB = 116, TR_V = 122, TR_VA = 134, TR_VB = 140;
  const int VREG = 146;

  // ---- workspace carve-up --------------------------------------------------
  size_t off = 0;
  auto alloc = [&](size_t nElem) -> float* {
    float* p = (float*)((char*)d_ws + off);
    off += (nElem * 4 + 255) & ~(size_t)255;
    return p;
  };
  float* XA0 = alloc((size_t)NA_ * 64);
  float* XT0 = alloc((size_t)NT_ * 64);
  float* XA1 = alloc((size_t)NA_ * 64);
  float* XT1 = alloc((size_t)NT_ * 64);
  float* HS  = alloc((size_t)NT_ * 256);
  float* HDm = alloc((size_t)NT_ * 256);
  float* accA = alloc((size_t)NA_ * 256);
  float* accT = alloc((size_t)NT_ * 256);
  float* asrc = alloc((size_t)NT_ * 4);
  float* adst = alloc((size_t)NT_ * 4);
  unsigned* amax = (unsigned*)alloc((size_t)NT_ * 4);
  float* denom = alloc((size_t)NT_ * 4);
  float* sbuf  = alloc((size_t)E_ * 4);
  float* stats = alloc(256);   // sumA, sqA, sumT, sqT (64 each)
  float* h1 = alloc((size_t)NT_ * 64);
  float* h2 = alloc((size_t)NT_ * 32);
  float* h3 = alloc((size_t)NT_ * 16);
  float* Ya = HS;                         // reuse after GEMMs are consumed
  float* Yt = HS + (size_t)NA_ * 64;

  auto gemm = [&](const float* A, const float* B, const float* bias, float* C,
                  int M, int N, int K, int act) {
    dim3 g((M + 63) / 64, (N + 63) / 64);
    gemm_wmma_kernel<<<g, 256, 0, stream>>>(A, B, bias, C, M, N, K, act);
  };

  // ---- embeddings ----------------------------------------------------------
  gemm(P(0), P(EMB_AW), P(EMB_AB), XA0, NA_, 64, 32, 1);
  gemm(P(1), P(EMB_TW), P(EMB_TB), XT0, NT_, 64, 48, 1);

  const int HEADS[3] = {4, 4, 2};
  float* XaIn = XA0; float* XtIn = XT0;
  float* XaOut = XA1; float* XtOut = XT1;

  for (int l = 0; l < 3; ++l) {
    const int H = HEADS[l];
    const int HD = H * 64;

    hipMemsetAsync(accA, 0, (size_t)NA_ * HD * 4, stream);
    hipMemsetAsync(accT, 0, (size_t)NT_ * HD * 4, stream);

    struct Rel { int slot; const float* Xs; const float* Xd; int Ns, Nd;
                 const int* es; const int* ed; float* acc; };
    const Rel rels[4] = {
      {0, XaIn, XtIn, NA_, NT_, I(2), I(3), accT},  // at1
      {1, XaIn, XtIn, NA_, NT_, I(6), I(7), accT},  // at2
      {2, XtIn, XaIn, NT_, NA_, I(4), I(5), accA},  // ta1
      {3, XtIn, XaIn, NT_, NA_, I(8), I(9), accA},  // ta2
    };

    for (int r = 0; r < 4; ++r) {
      const Rel& R = rels[r];
      int pb = CONV0 + l * 16 + R.slot * 4;   // W, att_dst, att_src, bias
      gemm(R.Xs, P(pb + 0), nullptr, HS,  R.Ns, HD, 64, 0);
      gemm(R.Xd, P(pb + 0), nullptr, HDm, R.Nd, HD, 64, 0);
      head_dot_kernel<<<(R.Ns * H + 255) / 256, 256, 0, stream>>>(HS,  P(pb + 2), asrc, R.Ns, H);
      head_dot_kernel<<<(R.Nd * H + 255) / 256, 256, 0, stream>>>(HDm, P(pb + 1), adst, R.Nd, H);
      hipMemsetAsync(amax,  0, (size_t)R.Nd * H * 4, stream);
      hipMemsetAsync(denom, 0, (size_t)R.Nd * H * 4, stream);
      edge_score_max_kernel<<<(E_ * H + 255) / 256, 256, 0, stream>>>(
          R.es, R.ed, asrc, adst, sbuf, amax, H);
      edge_exp_kernel<<<(E_ * H + 255) / 256, 256, 0, stream>>>(
          R.ed, sbuf, amax, denom, H);
      edge_scatter_kernel<<<(E_ * H * 16 + 255) / 256, 256, 0, stream>>>(
          R.es, R.ed, sbuf, denom, HS, R.acc, H);
    }

    // finalize: agent gets ta1+ta2 biases, track gets at1+at2 biases
    hipMemsetAsync(stats, 0, 256 * 4, stream);
    const float* b_ta1 = P(CONV0 + l * 16 + 2 * 4 + 3);
    const float* b_ta2 = P(CONV0 + l * 16 + 3 * 4 + 3);
    const float* b_at1 = P(CONV0 + l * 16 + 0 * 4 + 3);
    const float* b_at2 = P(CONV0 + l * 16 + 1 * 4 + 3);
    gat_finalize_kernel<<<(NA_ + 3) / 4, 256, 0, stream>>>(
        accA, b_ta1, b_ta2, Ya, stats + 0,  stats + 64, NA_, H);
    gat_finalize_kernel<<<(NT_ + 3) / 4, 256, 0, stream>>>(
        accT, b_at1, b_at2, Yt, stats + 128, stats + 192, NT_, H);

    const float* residA = (l > 0) ? XaIn : nullptr;
    const float* residT = (l > 0) ? XtIn : nullptr;
    bn_apply_kernel<<<(NA_ * 64 + 255) / 256, 256, 0, stream>>>(
        Ya, stats + 0,  stats + 64,  P(BN0 + l * 4 + 1), P(BN0 + l * 4 + 0),
        residA, XaOut, NA_);
    bn_apply_kernel<<<(NT_ * 64 + 255) / 256, 256, 0, stream>>>(
        Yt, stats + 128, stats + 192, P(BN0 + l * 4 + 3), P(BN0 + l * 4 + 2),
        residT, XtOut, NT_);

    float* t;
    t = XaIn; XaIn = XaOut; XaOut = t;
    t = XtIn; XtIn = XtOut; XtOut = t;
  }
  // after 3 swaps, final features are in XaIn / XtIn
  const float* Xfin[2] = {XaIn, XtIn};
  const int    Nfin[2] = {NA_, NT_};
  float*       Obase[2] = {(float*)d_out, (float*)d_out + (size_t)NA_ * 5};
  const int    PolBase[2][4] = {{AG_VA, AG_VB, AG_CA, AG_CB},
                                {TR_VA, TR_VB, TR_CA, TR_CB}};
  const int    ValBase[2] = {AG_V, TR_V};

  for (int t = 0; t < 2; ++t) {
    const float* X = Xfin[t];
    const int Nn = Nfin[t];
    float* ob = Obase[t];
    const int wblk = (Nn + 7) / 8;

    // four policy heads -> columns 0..3
    for (int hidx = 0; hidx < 4; ++hidx) {
      int pb = PolBase[t][hidx];
      gemm(X, P(pb + 0), P(pb + 2), h2, Nn, 32, 64, 0);
      ln_relu_kernel<<<wblk, 256, 0, stream>>>(h2, P(pb + 5), P(pb + 4), h2, Nn, 32);
      policy_out_kernel<<<wblk, 256, 0, stream>>>(h2, P(pb + 1), P(pb + 3), ob, Nn, hidx);
    }
    // value head -> column 4
    int vb = ValBase[t];
    gemm(X, P(vb + 0), P(vb + 4), h1, Nn, 64, 64, 0);
    ln_relu_kernel<<<wblk, 256, 0, stream>>>(h1, P(vb + 9), P(vb + 8), h1, Nn, 64);
    gemm(h1, P(vb + 1), P(vb + 5), h2, Nn, 32, 64, 0);
    ln_relu_kernel<<<wblk, 256, 0, stream>>>(h2, P(vb + 11), P(vb + 10), h2, Nn, 32);
    gemm(h2, P(vb + 2), P(vb + 6), h3, Nn, 16, 32, 1);
    value_out_kernel<<<wblk, 256, 0, stream>>>(h3, P(vb + 3), P(vb + 7), P(VREG), ob, Nn);
  }
}
